// WindowBasedGraphAttentionBlock_54262616818068
// MI455X (gfx1250) — matrix operand order repaired and checked by reading, not compiled, not run on hardware
//
#include <hip/hip_runtime.h>
#include <cstdint>
#include <cstddef>

// ---------------------------------------------------------------------------
// Types for CDNA5 WMMA (wave32): V_WMMA_F32_16X16X32_F16
// ---------------------------------------------------------------------------
typedef __attribute__((ext_vector_type(16))) _Float16 v16h;
typedef __attribute__((ext_vector_type(8)))  _Float16 v8h;
typedef __attribute__((ext_vector_type(8)))  float    v8f;

union F16x16 { v16h v; v8h h[2]; _Float16 e[16]; };

// A fragment (16x32, f16): lane l holds row m = l&15.
// lanes 0-15:  K = koff + [0..7]  (v0..v3) and koff + [16..23] (v4..v7)
// lanes 16-31: K = koff + [8..15]           and koff + [24..31]
__device__ __forceinline__ v16h load_a_frag(const _Float16* lds, int rowStride,
                                            int koff, int lane) {
  int m  = lane & 15;
  int kb = koff + ((lane >> 4) << 3);
  F16x16 f;
  f.h[0] = *(const v8h*)(lds + m * rowStride + kb);
  f.h[1] = *(const v8h*)(lds + m * rowStride + kb + 16);
  return f.v;
}

// B fragment (32x16, f16) from column-major staging Bt[n][k]:
// lane l holds column n = l&15; lanes 0-15: K = koff+[0..7],[16..23], lanes 16-31: +[8..15],[24..31]
__device__ __forceinline__ v16h load_b_frag(const _Float16* ldsBt, int rowStride,
                                            int koff, int lane) {
  int n  = lane & 15;
  int kb = koff + ((lane >> 4) << 3);
  F16x16 f;
  f.h[0] = *(const v8h*)(ldsBt + n * rowStride + kb);
  f.h[1] = *(const v8h*)(ldsBt + n * rowStride + kb + 16);
  return f.v;
}

// CDNA5 async global->LDS copy: 16 bytes per lane, tracked by ASYNCcnt.
__device__ __forceinline__ void async_copy16(uint32_t lds_byte, const void* gaddr) {
#if defined(__gfx1250__)
  asm volatile("global_load_async_to_lds_b128 %0, %1, off"
               :: "v"(lds_byte), "v"(gaddr)
               : "memory");
#endif
}
__device__ __forceinline__ void wait_async0() {
#if defined(__gfx1250__)
  asm volatile("s_wait_asynccnt 0" ::: "memory");
#endif
}

// ---------------------------------------------------------------------------
// Constants
// ---------------------------------------------------------------------------
#define NWIN   162
#define EMB    256
#define NHEAD  8
#define DHEAD  32
#define HIDDEN 1024
#define NVERT  40962
#define BATCH  2

// ---------------------------------------------------------------------------
// Kernel 0: one-time weight convert + transpose: W[K][Nf] f32 -> Wt[Nf][K] f16
// ---------------------------------------------------------------------------
__global__ void __launch_bounds__(256) wconv_kernel(const float* __restrict__ W,
                                                    _Float16* __restrict__ Wt,
                                                    int K, int Nf) {
  int e = blockIdx.x * 256 + threadIdx.x;
  if (e < K * Nf) {
    int k = e / Nf, n = e % Nf;
    Wt[(size_t)n * K + k] = (_Float16)W[e];
  }
}

// ---------------------------------------------------------------------------
// Kernel 1: gather windows + LayerNorm1 -> f16 h  (one block per slot row)
// ---------------------------------------------------------------------------
__global__ void __launch_bounds__(256) gather_ln1_kernel(
    const float* __restrict__ x, const int* __restrict__ gather_idx,
    const int* __restrict__ inv_idx, const float* __restrict__ gmm,
    const float* __restrict__ bta, _Float16* __restrict__ h16,
    int MW) {
  __shared__ float red[256];
  __shared__ float stat[2];
  const int s = blockIdx.x;                 // 0 .. B*NW*MW-1
  const int c = threadIdx.x;                // channel
  const int perBatch = NWIN * MW;
  const int slotInBatch = s % perBatch;     // i*MW + j
  const int b = s / perBatch;
  const int vtx = gather_idx[slotInBatch];
  const bool valid = (inv_idx[vtx] == slotInBatch);
  float val = valid ? x[((size_t)b * NVERT + vtx) * EMB + c] : 0.0f;

  red[c] = val; __syncthreads();
  for (int st = 128; st > 0; st >>= 1) { if (c < st) red[c] += red[c + st]; __syncthreads(); }
  if (c == 0) stat[0] = red[0] * (1.0f / EMB);
  __syncthreads();
  const float mean = stat[0];
  const float dv = val - mean;
  red[c] = dv * dv; __syncthreads();
  for (int st = 128; st > 0; st >>= 1) { if (c < st) red[c] += red[c + st]; __syncthreads(); }
  if (c == 0) stat[1] = red[0] * (1.0f / EMB);
  __syncthreads();
  const float var = stat[1];
  const float hv = dv * rsqrtf(var + 1e-5f) * gmm[c] + bta[c];
  h16[(size_t)s * EMB + c] = (_Float16)hv;
}

// ---------------------------------------------------------------------------
// Kernel 5: scatter + residual + LayerNorm2 -> x2 (f32), h2 (f16)
// ---------------------------------------------------------------------------
__global__ void __launch_bounds__(256) residual_ln2_kernel(
    const float* __restrict__ x, const float* __restrict__ proj,
    const int* __restrict__ inv_idx, const float* __restrict__ gmm,
    const float* __restrict__ bta, float* __restrict__ x2,
    _Float16* __restrict__ h2, int MW) {
  __shared__ float red[256];
  __shared__ float stat[2];
  const int r = blockIdx.x;                 // 0 .. B*N-1
  const int c = threadIdx.x;
  const int b = r / NVERT;
  const int n = r % NVERT;
  const int slot = inv_idx[n];
  float val = x[(size_t)r * EMB + c] +
              proj[((size_t)b * NWIN * MW + slot) * EMB + c];
  x2[(size_t)r * EMB + c] = val;

  red[c] = val; __syncthreads();
  for (int st = 128; st > 0; st >>= 1) { if (c < st) red[c] += red[c + st]; __syncthreads(); }
  if (c == 0) stat[0] = red[0] * (1.0f / EMB);
  __syncthreads();
  const float mean = stat[0];
  const float dv = val - mean;
  red[c] = dv * dv; __syncthreads();
  for (int st = 128; st > 0; st >>= 1) { if (c < st) red[c] += red[c + st]; __syncthreads(); }
  if (c == 0) stat[1] = red[0] * (1.0f / EMB);
  __syncthreads();
  const float var = stat[1];
  const float hv = dv * rsqrtf(var + 1e-5f) * gmm[c] + bta[c];
  h2[(size_t)r * EMB + c] = (_Float16)hv;
}

// ---------------------------------------------------------------------------
// WMMA GEMM: out[M,Nfull] = A[M,K](f16) @ Wt[Nfull,K](f16, pre-transposed) + bias
// block = 128 thr (4 waves), tile 64x64, K step 32, double-buffered LDS with
// async global->LDS staging (ASYNCcnt).  EPI: 0=none, 1=exact GELU.
// ---------------------------------------------------------------------------
template <int EPI, bool OUT_F16, bool RES>
__global__ void __launch_bounds__(128) gemm_kernel(
    const _Float16* __restrict__ A, const _Float16* __restrict__ Wt,
    const float* __restrict__ bias, const float* __restrict__ res,
    _Float16* __restrict__ out16, float* __restrict__ out32,
    int M, int K, int Nfull) {
  __shared__ __align__(16) _Float16 sA[2][64 * 32];   // [row][k]
  __shared__ __align__(16) _Float16 sBt[2][64 * 32];  // [n][k]
  const int tid  = threadIdx.x;
  const int lane = tid & 31;
  const int wave = tid >> 5;
  const int m0 = blockIdx.x * 64;
  const int n0 = blockIdx.y * 64;
  const bool interior = (m0 + 64 <= M);

  const uint32_t ldsA = (uint32_t)(uintptr_t)(&sA[0][0]);
  const uint32_t ldsB = (uint32_t)(uintptr_t)(&sBt[0][0]);

  auto stageA = [&](int buf, int k0) {
    if (interior) {
#pragma unroll
      for (int i = 0; i < 2; ++i) {
        int c = tid + 128 * i;                       // chunk 0..255 (16B each)
        const _Float16* g = A + (size_t)(m0 + (c >> 2)) * K + k0 + (c & 3) * 8;
        async_copy16(ldsA + (uint32_t)buf * 4096u + (uint32_t)c * 16u, g);
      }
    } else {
#pragma unroll
      for (int i = 0; i < 2; ++i) {
        int c = tid + 128 * i;
        int row = c >> 2, off = (c & 3) * 8;
        v8h val = {};
        int m = m0 + row;
        if (m < M) val = *(const v8h*)(A + (size_t)m * K + k0 + off);
        *(v8h*)(&sA[buf][row * 32 + off]) = val;
      }
    }
  };
  auto stageB = [&](int buf, int k0) {
#pragma unroll
    for (int i = 0; i < 2; ++i) {
      int c = tid + 128 * i;
      const _Float16* g = Wt + (size_t)(n0 + (c >> 2)) * K + k0 + (c & 3) * 8;
      async_copy16(ldsB + (uint32_t)buf * 4096u + (uint32_t)c * 16u, g);
    }
  };

  v8f acc[4] = {};
  const int nk = K >> 5;
  stageA(0, 0);
  stageB(0, 0);
  for (int kt = 0; kt < nk; ++kt) {
    const int buf = kt & 1;
    wait_async0();
    __syncthreads();
    if (kt + 1 < nk) {            // prefetch next K-tile into the other buffer
      stageA(1 - buf, (kt + 1) * 32);
      stageB(1 - buf, (kt + 1) * 32);
    }
    v16h a = load_a_frag(&sA[buf][0] + wave * 16 * 32, 32, 0, lane);
#pragma unroll
    for (int nt = 0; nt < 4; ++nt) {
      v16h b = load_b_frag(&sBt[buf][0] + nt * 16 * 32, 32, 0, lane);
      acc[nt] = __builtin_amdgcn_wmma_f32_16x16x32_f16(
          false, a, false, b, (short)0, acc[nt], false, false);
    }
    __syncthreads();
  }

  // epilogue: lane l -> n = l&15, rows m = wave*16 + (l>>4)*8 + r
  const int mBase = m0 + wave * 16 + ((lane >> 4) << 3);
  const int nl = lane & 15;
#pragma unroll
  for (int nt = 0; nt < 4; ++nt) {
    const int n = n0 + nt * 16 + nl;
    const float bv = bias[n];
#pragma unroll
    for (int r = 0; r < 8; ++r) {
      const int m = mBase + r;
      if (m >= M) continue;
      float v = acc[nt][r] + bv;
      if (EPI == 1) v = 0.5f * v * (1.0f + erff(v * 0.70710678118654752f));
      if (RES) v += res[(size_t)m * Nfull + n];
      if (OUT_F16) out16[(size_t)m * Nfull + n] = (_Float16)v;
      else         out32[(size_t)m * Nfull + n] = v;
    }
  }
}

// ---------------------------------------------------------------------------
// Kernel 3: windowed multi-head attention
// grid = (ceil(MW/16), B*NW, H); block = 128 (4 waves); dynamic LDS.
// S = q k^T * scale + bias (WMMA, K=Dh=32), fp32 softmax, O = P V (WMMA).
// ---------------------------------------------------------------------------
__global__ void __launch_bounds__(128) attn_kernel(
    const _Float16* __restrict__ qkv, const float* __restrict__ rel_bias,
    _Float16* __restrict__ attn_out, int MW, int MWpad) {
  extern __shared__ char smem_raw[];
  _Float16* sK  = (_Float16*)smem_raw;          // [MWpad][32]  keys
  _Float16* sVt = sK  + MWpad * 32;             // [32][MWpad]  values^T
  _Float16* sQ  = sVt + MWpad * 32;             // [16][32]     query tile
  float*    sS  = (float*)(sQ + 16 * 32);       // [16][MWpad]  scores
  _Float16* sP  = (_Float16*)(sS + 16 * MWpad); // [16][MWpad]  probs f16

  const int tid  = threadIdx.x;
  const int lane = tid & 31;
  const int wave = tid >> 5;
  const int q0   = blockIdx.x * 16;
  const int bw   = blockIdx.y;                  // window (b*NW + i)
  const int head = blockIdx.z;
  const size_t rowBase = (size_t)bw * MW;
  const float scale = 0.17677669529663688f;     // 1/sqrt(32)
  const float rbias = rel_bias[head];

  // stage K and V^T
  for (int idx = tid; idx < MWpad * 32; idx += 128) {
    int j = idx >> 5, d = idx & 31;
    _Float16 kv = (_Float16)0, vv = (_Float16)0;
    if (j < MW) {
      const _Float16* row = qkv + (rowBase + j) * 768 + head * DHEAD;
      kv = row[256 + d];
      vv = row[512 + d];
    }
    sK[j * 32 + d] = kv;
    sVt[d * MWpad + j] = vv;
  }
  // stage Q tile
  for (int idx = tid; idx < 16 * 32; idx += 128) {
    int m = idx >> 5, d = idx & 31;
    int j = q0 + m;
    _Float16 qv = (_Float16)0;
    if (j < MW) qv = qkv[(rowBase + j) * 768 + head * DHEAD + d];
    sQ[m * 32 + d] = qv;
  }
  __syncthreads();

  // S = q k^T (one WMMA per 16-key tile; sK rows are B columns)
  const v16h a = load_a_frag(sQ, 32, 0, lane);
  const int ntiles = MWpad >> 4;
  for (int t = wave; t < ntiles; t += 4) {
    v16h b = load_b_frag(sK + t * 16 * 32, 32, 0, lane);
    v8f c = {};
    c = __builtin_amdgcn_wmma_f32_16x16x32_f16(false, a, false, b, (short)0, c,
                                               false, false);
    const int n  = t * 16 + (lane & 15);
    const int mb = (lane >> 4) << 3;
#pragma unroll
    for (int r = 0; r < 8; ++r) sS[(mb + r) * MWpad + n] = c[r] * scale + rbias;
  }
  __syncthreads();

  // fp32 softmax over the MW keys; write P (f16), zero the pad
  if (tid < 16) {
    float* row = sS + tid * MWpad;
    float mx = -1e30f;
    for (int j = 0; j < MW; ++j) mx = fmaxf(mx, row[j]);
    float sum = 0.0f;
    for (int j = 0; j < MW; ++j) { float e = __expf(row[j] - mx); row[j] = e; sum += e; }
    const float inv = 1.0f / sum;
    _Float16* prow = sP + tid * MWpad;
    for (int j = 0; j < MW; ++j)      prow[j] = (_Float16)(row[j] * inv);
    for (int j = MW; j < MWpad; ++j)  prow[j] = (_Float16)0;
  }
  __syncthreads();

  // O = P @ V : waves 0,1 each own a 16-wide d-tile, K loop over MWpad
  if (wave < 2) {
    v8f acc = {};
    for (int kt = 0; kt < MWpad; kt += 32) {
      v16h pa = load_a_frag(sP, MWpad, kt, lane);
      v16h vb = load_b_frag(sVt + wave * 16 * MWpad, MWpad, kt, lane);
      acc = __builtin_amdgcn_wmma_f32_16x16x32_f16(false, pa, false, vb,
                                                   (short)0, acc, false, false);
    }
    const int nl = lane & 15;
    const int mb = (lane >> 4) << 3;
#pragma unroll
    for (int r = 0; r < 8; ++r) {
      const int m = q0 + mb + r;
      if (m < MW)
        attn_out[(rowBase + m) * EMB + head * DHEAD + wave * 16 + nl] =
            (_Float16)acc[r];
    }
  }
}

// ---------------------------------------------------------------------------
// Kernel 8: copy mesh to output tail
// ---------------------------------------------------------------------------
__global__ void __launch_bounds__(256) copy_kernel(const float* __restrict__ s,
                                                   float* __restrict__ d, int n) {
  int i = blockIdx.x * 256 + threadIdx.x;
  if (i < n) d[i] = s[i];
}

// ---------------------------------------------------------------------------
// Host launcher
// ---------------------------------------------------------------------------
extern "C" void kernel_launch(void* const* d_in, const int* in_sizes, int n_in,
                              void* d_out, int out_size, void* d_ws,
                              size_t ws_size, hipStream_t stream) {
  const float* x        = (const float*)d_in[0];
  const float* mesh     = (const float*)d_in[1];
  const float* w_qkv    = (const float*)d_in[2];
  const float* b_qkv    = (const float*)d_in[3];
  const float* rel_bias = (const float*)d_in[4];
  const float* w_proj   = (const float*)d_in[5];
  const float* b_proj   = (const float*)d_in[6];
  const float* ln1_g    = (const float*)d_in[7];
  const float* ln1_b    = (const float*)d_in[8];
  const float* ln2_g    = (const float*)d_in[9];
  const float* ln2_b    = (const float*)d_in[10];
  const float* w1       = (const float*)d_in[11];
  const float* b1       = (const float*)d_in[12];
  const float* w2       = (const float*)d_in[13];
  const float* b2       = (const float*)d_in[14];
  const int*   gidx     = (const int*)d_in[15];
  const int*   inv_idx  = (const int*)d_in[17];

  const int MW    = in_sizes[15] / NWIN;
  const int MWpad = (MW + 31) & ~31;
  const int Mtot  = BATCH * NWIN * MW;     // window-slot rows
  const int BN    = BATCH * NVERT;         // vertex rows

  // workspace carve (256B aligned), with lifetime-based aliasing
  char* p = (char*)d_ws;
  auto carve = [&](size_t bytes) {
    void* r = (void*)p;
    p += (bytes + 255) & ~(size_t)255;
    return r;
  };
  size_t rows1 = (size_t)(Mtot > BN ? Mtot : BN);
  _Float16* buf1 = (_Float16*)carve(rows1 * EMB * 2);          // h16 / attn16 / h2
  char*     buf2 = (char*)carve((size_t)Mtot * 768 * 2);       // qkv16, later proj32
  float*    x2   = (float*)carve((size_t)BN * EMB * 4);        // residual stream
  _Float16* mid  = (_Float16*)carve((size_t)BN * HIDDEN * 2);  // MLP hidden
  _Float16* wq16 = (_Float16*)carve((size_t)EMB * 768 * 2);    // f16 transposed weights
  _Float16* wp16 = (_Float16*)carve((size_t)EMB * EMB * 2);
  _Float16* w1t  = (_Float16*)carve((size_t)EMB * HIDDEN * 2);
  _Float16* w2t  = (_Float16*)carve((size_t)HIDDEN * EMB * 2);

  _Float16* qkv16  = (_Float16*)buf2;
  float*    proj32 = (float*)buf2;
  float*    out32  = (float*)d_out;

  // 0) one-time weight convert + transpose to f16
  wconv_kernel<<<(EMB * 768 + 255) / 256, 256, 0, stream>>>(w_qkv, wq16, EMB, 768);
  wconv_kernel<<<(EMB * EMB + 255) / 256, 256, 0, stream>>>(w_proj, wp16, EMB, EMB);
  wconv_kernel<<<(EMB * HIDDEN + 255) / 256, 256, 0, stream>>>(w1, w1t, EMB, HIDDEN);
  wconv_kernel<<<(HIDDEN * EMB + 255) / 256, 256, 0, stream>>>(w2, w2t, HIDDEN, EMB);

  // 1) gather + LN1
  gather_ln1_kernel<<<Mtot, 256, 0, stream>>>(x, gidx, inv_idx, ln1_g, ln1_b,
                                              buf1, MW);
  // 2) QKV GEMM (f16 out)
  gemm_kernel<0, true, false><<<dim3((Mtot + 63) / 64, 768 / 64), 128, 0,
                                stream>>>(buf1, wq16, b_qkv, nullptr, qkv16,
                                          nullptr, Mtot, EMB, 768);
  // 3) attention (dynamic LDS)
  size_t shbytes = (size_t)MWpad * 224 + 1024;
  hipFuncSetAttribute((const void*)attn_kernel,
                      hipFuncAttributeMaxDynamicSharedMemorySize, (int)shbytes);
  attn_kernel<<<dim3((MW + 15) / 16, BATCH * NWIN, NHEAD), 128, shbytes,
                stream>>>(qkv16, rel_bias, buf1, MW, MWpad);
  // 4) output projection (f32 out)
  gemm_kernel<0, false, false><<<dim3((Mtot + 63) / 64, EMB / 64), 128, 0,
                                 stream>>>(buf1, wp16, b_proj, nullptr,
                                           nullptr, proj32, Mtot, EMB, EMB);
  // 5) scatter + residual + LN2
  residual_ln2_kernel<<<BN, 256, 0, stream>>>(x, proj32, inv_idx, ln2_g, ln2_b,
                                              x2, buf1, MW);
  // 6) MLP fc1 + exact GELU (f16 out)
  gemm_kernel<1, true, false><<<dim3((BN + 63) / 64, HIDDEN / 64), 128, 0,
                                stream>>>(buf1, w1t, b1, nullptr, mid, nullptr,
                                          BN, EMB, HIDDEN);
  // 7) MLP fc2 + residual -> final output
  gemm_kernel<0, false, true><<<dim3((BN + 63) / 64, EMB / 64), 128, 0,
                                stream>>>(mid, w2t, b2, x2, nullptr, out32, BN,
                                          HIDDEN, EMB);
  // 8) mesh passthrough
  const int meshN = NVERT * 3;
  copy_kernel<<<(meshN + 255) / 256, 256, 0, stream>>>(
      mesh, out32 + (size_t)BN * EMB, meshN);
}
